// LinearAttentionTransformer_57629871178280
// MI455X (gfx1250) — matrix operand order repaired and checked by reading, not compiled, not run on hardware
//
#include <hip/hip_runtime.h>
#include <math.h>

typedef __bf16 bf16;
typedef __attribute__((ext_vector_type(2)))  __bf16 bf16x2;
typedef __attribute__((ext_vector_type(4)))  __bf16 bf16x4;
typedef __attribute__((ext_vector_type(8)))  __bf16 bf16x8;
typedef __attribute__((ext_vector_type(16))) __bf16 v16bf;
typedef __attribute__((ext_vector_type(8)))  float  v8f;

#define TILE_M 128
#define TILE_N 128
#define TILE_K 64    // two 16x16x32 WMMA K-steps per staged tile
#define LDSS   72    // padded LDS row stride (elems); 144B rows keep 16B alignment

// ---------------------------------------------------------------------------
// Stage one 128x64 A tile (As[m][k]) and 64x128 B tile (Bs[n][k], transposed)
// into LDS. 16-byte global loads; 256 threads.
// transA=1: A stored [K x M] -> scatter-transpose like B.
// ---------------------------------------------------------------------------
__device__ __forceinline__
void load_tile(const bf16* __restrict__ A, const bf16* __restrict__ B,
               bf16* __restrict__ As, bf16* __restrict__ Bs,
               int k0, int blockM, int blockN, int lda, int ldb,
               int transA, int tid)
{
    if (!transA) {
        #pragma unroll
        for (int c = 0; c < 4; ++c) {
            int chunk = tid + 256 * c;        // 1024 chunks of 8 elems
            int row   = chunk >> 3;           // 0..127
            int kc    = (chunk & 7) << 3;     // 0..56
            bf16x8 v = *reinterpret_cast<const bf16x8*>(
                A + (long)(blockM + row) * lda + (k0 + kc));
            *reinterpret_cast<bf16x8*>(&As[row * LDSS + kc]) = v;
        }
    } else {
        #pragma unroll
        for (int c = 0; c < 4; ++c) {
            int chunk = tid + 256 * c;
            int krow  = chunk >> 4;           // 0..63
            int mc    = (chunk & 15) << 3;    // 0..120
            bf16x8 v = *reinterpret_cast<const bf16x8*>(
                A + (long)(k0 + krow) * lda + (blockM + mc));
            #pragma unroll
            for (int j = 0; j < 8; ++j)
                As[(mc + j) * LDSS + krow] = v[j];
        }
    }
    #pragma unroll
    for (int c = 0; c < 4; ++c) {
        int chunk = tid + 256 * c;
        int krow  = chunk >> 4;               // 0..63
        int nc    = (chunk & 15) << 3;        // 0..120
        bf16x8 v = *reinterpret_cast<const bf16x8*>(
            B + (long)(k0 + krow) * ldb + (blockN + nc));
        #pragma unroll
        for (int j = 0; j < 8; ++j)
            Bs[(nc + j) * LDSS + krow] = v[j];
    }
}

// ---------------------------------------------------------------------------
// Generic bf16 WMMA GEMM: C(f32 accum) = op(A) x B, tiles 128x128x64,
// LDS double-buffered. 256 threads = 8 waves; wave grid 4(M) x 2(N);
// each wave owns 2x4 16x16 fragments -> 16 v_wmma per iteration.
// mode 0: split-store qkv (q->f32, k->f32, v->bf16), widths N/3 each
// mode 1/2: store bf16 (batched via blockIdx.z strides)
// mode 3: store f32 with +bias[col] +residual[row][col]
// All M,N multiples of 128 and K multiples of 64 (guaranteed by launcher).
// ---------------------------------------------------------------------------
__global__ __launch_bounds__(256)
void gemm_bf16_wmma(const bf16* __restrict__ A, const bf16* __restrict__ B,
                    int M, int N, int K, int lda, int ldb, int transA,
                    long sA, long sB, long sC, int mode,
                    float* __restrict__ out_q, float* __restrict__ out_k,
                    bf16* __restrict__ out_v,
                    bf16* __restrict__ out_bf, int ldc_bf,
                    float* __restrict__ out_f, int ldc_f,
                    const float* __restrict__ bias,
                    const float* __restrict__ residual, int ldres)
{
    __shared__ bf16 As[2][TILE_M * LDSS];
    __shared__ bf16 Bs[2][TILE_N * LDSS];

    const int z = blockIdx.z;
    A += (long)z * sA;
    B += (long)z * sB;

    const int tid  = threadIdx.x;
    const int lane = tid & 31;
    const int wave = tid >> 5;
    const int wm   = wave & 3;   // 4 M-groups of 32 rows
    const int wn   = wave >> 2;  // 2 N-groups of 64 cols
    const int hl   = lane >> 4;  // lane half (0/1)
    const int l16  = lane & 15;

    const int blockM = blockIdx.x * TILE_M;
    const int blockN = blockIdx.y * TILE_N;

    v8f acc[2][4];
    #pragma unroll
    for (int i = 0; i < 2; ++i)
        #pragma unroll
        for (int j = 0; j < 4; ++j)
            #pragma unroll
            for (int r = 0; r < 8; ++r)
                acc[i][j][r] = 0.0f;

    const int nIter = K / TILE_K;
    load_tile(A, B, As[0], Bs[0], 0, blockM, blockN, lda, ldb, transA, tid);
    __syncthreads();

    for (int it = 0; it < nIter; ++it) {
        const int cur = it & 1;
        if (it + 1 < nIter)
            load_tile(A, B, As[1 - cur], Bs[1 - cur], (it + 1) * TILE_K,
                      blockM, blockN, lda, ldb, transA, tid);
        if (it + 2 < nIter) {  // warm L2/WGP$ two tiles ahead
            const int kp = (it + 2) * TILE_K;
            __builtin_prefetch(A + (long)(blockM + (tid & 127)) * lda + kp, 0, 1);
            __builtin_prefetch(B + (long)(kp + (tid & 63)) * ldb + blockN +
                                   ((tid >> 6) << 5), 0, 1);
        }

        const bf16* as = As[cur];
        const bf16* bs = Bs[cur];
        #pragma unroll
        for (int ks = 0; ks < 2; ++ks) {
            // ---- fragments (ISA §7.12.2 layouts, wave32) ----
            // A 16x32: lanes0-15 M=lane; elem pairs (2v,2v+1):
            //          K = hl*8+2v (v<4), K = 16+hl*8+2(v-4) (v>=4)
            v16bf afrag[2];
            #pragma unroll
            for (int fm = 0; fm < 2; ++fm) {
                const bf16* ab = &as[(wm * 32 + fm * 16 + l16) * LDSS + ks * 32];
                v16bf a;
                #pragma unroll
                for (int vg = 0; vg < 4; ++vg) {
                    bf16x2 p = *reinterpret_cast<const bf16x2*>(ab + hl * 8 + 2 * vg);
                    a[2 * vg]     = p[0];
                    a[2 * vg + 1] = p[1];
                }
                #pragma unroll
                for (int vg = 0; vg < 4; ++vg) {
                    bf16x2 p = *reinterpret_cast<const bf16x2*>(ab + 16 + hl * 8 + 2 * vg);
                    a[8 + 2 * vg] = p[0];
                    a[9 + 2 * vg] = p[1];
                }
                afrag[fm] = a;
            }
            // B 32x16: lane -> N=l16; elements e<8 -> K = hl*8 + e, e>=8 -> K = 16 + hl*8 + (e-8)
            v16bf bfrag[4];
            #pragma unroll
            for (int fn = 0; fn < 4; ++fn) {
                const bf16* bb = &bs[(wn * 64 + fn * 16 + l16) * LDSS + ks * 32];
                v16bf bv;
                #pragma unroll
                for (int q4 = 0; q4 < 4; ++q4) {
                    bf16x4 p = *reinterpret_cast<const bf16x4*>(bb + (q4 >> 1) * 16 + hl * 8 + (q4 & 1) * 4);
                    bv[4 * q4]     = p[0];
                    bv[4 * q4 + 1] = p[1];
                    bv[4 * q4 + 2] = p[2];
                    bv[4 * q4 + 3] = p[3];
                }
                bfrag[fn] = bv;
            }
            #pragma unroll
            for (int fm = 0; fm < 2; ++fm)
                #pragma unroll
                for (int fn = 0; fn < 4; ++fn)
                    acc[fm][fn] = __builtin_amdgcn_wmma_f32_16x16x32_bf16(
                        false, afrag[fm], false, bfrag[fn],
                        (short)0, acc[fm][fn], false, false);
        }
        __syncthreads();
    }

    // ---- epilogue; C/D layout: VGPR r -> M = hl*8 + r, N = l16 ----
    const int dsub = N / 3;  // only used by mode 0
    #pragma unroll
    for (int fm = 0; fm < 2; ++fm) {
        #pragma unroll
        for (int fn = 0; fn < 4; ++fn) {
            const int rb  = blockM + wm * 32 + fm * 16 + hl * 8;
            const int col = blockN + wn * 64 + fn * 16 + l16;
            #pragma unroll
            for (int r = 0; r < 8; ++r) {
                const int row = rb + r;
                const float val = acc[fm][fn][r];
                if (mode == 0) {
                    if (col < dsub)
                        out_q[(long)row * dsub + col] = val;
                    else if (col < 2 * dsub)
                        out_k[(long)row * dsub + (col - dsub)] = val;
                    else
                        out_v[(long)row * dsub + (col - 2 * dsub)] = (bf16)val;
                } else if (mode == 3) {
                    out_f[(long)row * ldc_f + col] =
                        val + bias[col] + residual[(long)row * ldres + col];
                } else {
                    bf16* o = out_bf + (long)z * sC;
                    o[(long)row * ldc_bf + col] = (bf16)val;
                }
            }
        }
    }
}

// ---------------------------------------------------------------------------
// Block reductions (256 threads)
// ---------------------------------------------------------------------------
__device__ __forceinline__ float block_sum256(float v, float* sh) {
    int t = threadIdx.x;
    sh[t] = v; __syncthreads();
    for (int s = 128; s > 0; s >>= 1) {
        if (t < s) sh[t] += sh[t + s];
        __syncthreads();
    }
    float r = sh[0]; __syncthreads();
    return r;
}
__device__ __forceinline__ float block_max256(float v, float* sh) {
    int t = threadIdx.x;
    sh[t] = v; __syncthreads();
    for (int s = 128; s > 0; s >>= 1) {
        if (t < s) sh[t] = fmaxf(sh[t], sh[t + s]);
        __syncthreads();
    }
    float r = sh[0]; __syncthreads();
    return r;
}

// ---------------------------------------------------------------------------
// LayerNorm over last dim (d=1024), bf16 output
// ---------------------------------------------------------------------------
__global__ __launch_bounds__(256)
void ln_kernel(const float* __restrict__ x, const float* __restrict__ gamma,
               const float* __restrict__ beta, bf16* __restrict__ h, int d)
{
    __shared__ float sh[256];
    const long row = blockIdx.x;
    const float* xr = x + row * d;
    float s = 0.f;
    for (int i = threadIdx.x; i < d; i += 256) s += xr[i];
    const float mu = block_sum256(s, sh) / d;
    float v = 0.f;
    for (int i = threadIdx.x; i < d; i += 256) { float t = xr[i] - mu; v += t * t; }
    const float var = block_sum256(v, sh) / d;
    const float inv = rsqrtf(var + 1e-5f);
    bf16* hr = h + row * d;
    for (int i = threadIdx.x; i < d; i += 256)
        hr[i] = (bf16)((xr[i] - mu) * inv * gamma[i] + beta[i]);
}

// ---------------------------------------------------------------------------
// Row softmax over d (K path: softmax over channels), bf16 output
// ---------------------------------------------------------------------------
__global__ __launch_bounds__(256)
void softmax_row_kernel(const float* __restrict__ in, bf16* __restrict__ out, int d)
{
    __shared__ float sh[256];
    const long row = blockIdx.x;
    const float* r = in + row * d;
    float m = -INFINITY;
    for (int i = threadIdx.x; i < d; i += 256) m = fmaxf(m, r[i]);
    m = block_max256(m, sh);
    float s = 0.f;
    for (int i = threadIdx.x; i < d; i += 256) s += __expf(r[i] - m);
    s = block_sum256(s, sh);
    const float rs = 1.f / s;
    bf16* o = out + row * d;
    for (int i = threadIdx.x; i < d; i += 256) o[i] = (bf16)(__expf(r[i] - m) * rs);
}

// ---------------------------------------------------------------------------
// Column softmax over n (Q path: softmax over sequence). Block = (64 ch, 4).
// Coalesced: consecutive tx -> consecutive channels in memory.
// ---------------------------------------------------------------------------
__global__ void softmax_col_kernel(const float* __restrict__ in,
                                   bf16* __restrict__ out, int n, int d)
{
    __shared__ float sh[4][64];
    const int tx = threadIdx.x, ty = threadIdx.y;
    const int c = blockIdx.x * 64 + tx;
    const long base = (long)blockIdx.y * n * d + c;
    float m = -INFINITY;
    for (int i = ty; i < n; i += 4) m = fmaxf(m, in[base + (long)i * d]);
    sh[ty][tx] = m; __syncthreads();
    m = fmaxf(fmaxf(sh[0][tx], sh[1][tx]), fmaxf(sh[2][tx], sh[3][tx]));
    __syncthreads();
    float s = 0.f;
    for (int i = ty; i < n; i += 4) s += __expf(in[base + (long)i * d] - m);
    sh[ty][tx] = s; __syncthreads();
    s = sh[0][tx] + sh[1][tx] + sh[2][tx] + sh[3][tx];
    const float rs = 1.f / s;
    for (int i = ty; i < n; i += 4)
        out[base + (long)i * d] = (bf16)(__expf(in[base + (long)i * d] - m) * rs);
}

// ---------------------------------------------------------------------------
// fp32 -> bf16 elementwise (one-time weight conversion)
// ---------------------------------------------------------------------------
__global__ void f32_to_bf16_kernel(const float* __restrict__ in,
                                   bf16* __restrict__ out, int count)
{
    int i = blockIdx.x * blockDim.x + threadIdx.x;
    const int stride = gridDim.x * blockDim.x;
    for (; i < count; i += stride) out[i] = (bf16)in[i];
}

// ---------------------------------------------------------------------------
extern "C" void kernel_launch(void* const* d_in, const int* in_sizes, int n_in,
                              void* d_out, int out_size, void* d_ws, size_t ws_size,
                              hipStream_t stream)
{
    (void)in_sizes; (void)n_in; (void)out_size;
    const int Bb = 4, NT = 2048, D = 1024, DEPTH = 6;
    const int ROWS = Bb * NT;  // 8192

    const float* x_in  = (const float*)d_in[0];
    const float* w_qkv = (const float*)d_in[1];
    const float* w_out = (const float*)d_in[2];
    const float* b_out = (const float*)d_in[3];
    const float* gamma = (const float*)d_in[4];
    const float* beta  = (const float*)d_in[5];
    float* y = (float*)d_out;

    char* base = (char*)d_ws;
    size_t off = 0;
    auto take = [&](size_t bytes) -> char* {
        char* p = base + off;
        off += (bytes + 255) & ~(size_t)255;
        return p;
    };
    bf16*  wqkv_bf = (bf16*) take((size_t)DEPTH * D * 3 * D * sizeof(bf16));
    bf16*  wout_bf = (bf16*) take((size_t)DEPTH * D * D * sizeof(bf16));
    bf16*  h_bf    = (bf16*) take((size_t)ROWS * D * sizeof(bf16));
    float* q_f32   = (float*)take((size_t)ROWS * D * sizeof(float));
    float* k_f32   = (float*)take((size_t)ROWS * D * sizeof(float));
    bf16*  v_bf    = (bf16*) take((size_t)ROWS * D * sizeof(bf16));
    bf16*  qs_bf   = (bf16*) take((size_t)ROWS * D * sizeof(bf16));
    bf16*  ks_bf   = (bf16*) take((size_t)ROWS * D * sizeof(bf16));
    bf16*  S_bf    = (bf16*) take((size_t)Bb * D * D * sizeof(bf16));
    bf16*  attn_bf = (bf16*) take((size_t)ROWS * D * sizeof(bf16));
    float* x_buf   = (float*)take((size_t)ROWS * D * sizeof(float));
    if (off > ws_size) return;  // insufficient scratch: bail (avoid OOB)

    f32_to_bf16_kernel<<<2048, 256, 0, stream>>>(w_qkv, wqkv_bf, DEPTH * D * 3 * D);
    f32_to_bf16_kernel<<<2048, 256, 0, stream>>>(w_out, wout_bf, DEPTH * D * D);

    for (int l = 0; l < DEPTH; ++l) {
        const float* xcur = (l == 0) ? x_in : x_buf;
        float* xnext = (l == DEPTH - 1) ? y : x_buf;

        ln_kernel<<<ROWS, 256, 0, stream>>>(xcur, gamma + l * D, beta + l * D, h_bf, D);

        // qkv = h @ Wqkv : [8192 x 3072], split epilogue -> q/k fp32, v bf16
        gemm_bf16_wmma<<<dim3(ROWS / 128, (3 * D) / 128, 1), 256, 0, stream>>>(
            h_bf, wqkv_bf + (size_t)l * D * 3 * D,
            ROWS, 3 * D, D, D, 3 * D, /*transA=*/0,
            0L, 0L, 0L, /*mode=*/0,
            q_f32, k_f32, v_bf,
            nullptr, 0, nullptr, 0, nullptr, nullptr, 0);

        softmax_col_kernel<<<dim3(D / 64, Bb), dim3(64, 4), 0, stream>>>(q_f32, qs_bf, NT, D);
        softmax_row_kernel<<<ROWS, 256, 0, stream>>>(k_f32, ks_bf, D);

        // S = K^T Q per batch : [1024 x 1024], reduction over n=2048
        gemm_bf16_wmma<<<dim3(D / 128, D / 128, Bb), 256, 0, stream>>>(
            ks_bf, qs_bf,
            D, D, NT, D, D, /*transA=*/1,
            (long)NT * D, (long)NT * D, (long)D * D, /*mode=*/1,
            nullptr, nullptr, nullptr,
            S_bf, D, nullptr, 0, nullptr, nullptr, 0);

        // attn = V @ S per batch : [2048 x 1024]
        gemm_bf16_wmma<<<dim3(NT / 128, D / 128, Bb), 256, 0, stream>>>(
            v_bf, S_bf,
            NT, D, D, D, D, /*transA=*/0,
            (long)NT * D, (long)D * D, (long)NT * D, /*mode=*/2,
            nullptr, nullptr, nullptr,
            attn_bf, D, nullptr, 0, nullptr, nullptr, 0);

        // x_next = attn @ Wout + b_out + residual : [8192 x 1024] fp32
        gemm_bf16_wmma<<<dim3(ROWS / 128, D / 128, 1), 256, 0, stream>>>(
            attn_bf, wout_bf + (size_t)l * D * D,
            ROWS, D, D, D, D, /*transA=*/0,
            0L, 0L, 0L, /*mode=*/3,
            nullptr, nullptr, nullptr,
            nullptr, 0, xnext, D, b_out + l * D, xcur, D);
    }
}
